// GPTFromScratch_1606317768887
// MI455X (gfx1250) — matrix operand order repaired and checked by reading, not compiled, not run on hardware
//
#include <hip/hip_runtime.h>
#include <hip/hip_bf16.h>
#include <stdint.h>

#define V_  65
#define E_  384
#define T_  64
#define H_  6
#define D_  64
#define NL_ 4
#define FF_ (4*E_)
#define B_  1024
#define BT_ (B_*T_)

typedef __attribute__((ext_vector_type(16))) __bf16 v16bf;
typedef __attribute__((ext_vector_type(8)))  float  v8f;
typedef unsigned short u16b;

union FragAB { uint4 q[2]; u16b s[16]; v16bf v; };

// Scalar f32 -> bf16 (round-half-up).
__device__ __forceinline__ u16b f2bf(float f) {
  unsigned u = __builtin_bit_cast(unsigned, f) + 0x8000u;
  return (u16b)(u >> 16);
}

// Pack two f32 -> two bf16 in one dword via v_perm_b32 (1 perm + 2 adds).
__device__ __forceinline__ unsigned pkbf(float lo, float hi) {
  unsigned a = __builtin_bit_cast(unsigned, lo) + 0x8000u;
  unsigned b = __builtin_bit_cast(unsigned, hi) + 0x8000u;
  // result bytes: [0,1] = a[2,3], [2,3] = b[2,3]  (s1 supplies bytes 0-3)
  return __builtin_amdgcn_perm(b, a, 0x07060302u);
}

// Convert 16 consecutive f32 (4 float4) -> 16 bf16, store as 2 uint4 into LDS.
__device__ __forceinline__ void cvt16_store(const float4* __restrict__ s4, u16b* dst16) {
  float4 f0 = s4[0], f1 = s4[1], f2 = s4[2], f3 = s4[3];
  uint4 o0, o1;
  o0.x = pkbf(f0.x, f0.y); o0.y = pkbf(f0.z, f0.w);
  o0.z = pkbf(f1.x, f1.y); o0.w = pkbf(f1.z, f1.w);
  o1.x = pkbf(f2.x, f2.y); o1.y = pkbf(f2.z, f2.w);
  o1.z = pkbf(f3.x, f3.y); o1.w = pkbf(f3.z, f3.w);
  uint4* d = (uint4*)dst16;
  d[0] = o0; d[1] = o1;
}

__device__ __forceinline__ v8f wmma_bf16(const FragAB& a, const FragAB& b, v8f c) {
  return __builtin_amdgcn_wmma_f32_16x16x32_bf16(
      /*neg_a=*/false, a.v, /*neg_b=*/false, b.v,
      /*c_mod=*/(short)0, c, /*reuse_a=*/false, /*reuse_b=*/false);
}

// ---------------------------------------------------------------------------
// Weight convert: W f32 [K,N] row-major -> Wt bf16 [N,K] row-major (B-operand
// friendly: contiguous along K for a fixed output column n).
// ---------------------------------------------------------------------------
__global__ void wconv_kernel(const float* __restrict__ W, u16b* __restrict__ Wt,
                             int K, int N) {
  size_t i = (size_t)blockIdx.x * blockDim.x + threadIdx.x;
  if (i >= (size_t)K * N) return;
  int k = (int)(i % K);
  size_t n = i / K;
  Wt[n * K + k] = f2bf(W[(size_t)k * N + n]);
}

// ---------------------------------------------------------------------------
// Embedding: x[bt,e] = tok[idx[bt],e] + pos[t,e]
// ---------------------------------------------------------------------------
__global__ void embed_kernel(const int* __restrict__ idx, const float* __restrict__ tok,
                             const float* __restrict__ pos, float* __restrict__ X) {
  size_t i = (size_t)blockIdx.x * blockDim.x + threadIdx.x;
  if (i >= (size_t)BT_ * E_) return;
  int e = (int)(i % E_);
  size_t bt = i / E_;
  int t = (int)(bt % T_);
  X[i] = tok[(size_t)idx[bt] * E_ + e] + pos[(size_t)t * E_ + e];
}

// ---------------------------------------------------------------------------
// LayerNorm: one wave (32 lanes) per row of E=384 (12 elems/lane).
// ---------------------------------------------------------------------------
__global__ __launch_bounds__(256)
void layernorm_kernel(const float* __restrict__ X, const float* __restrict__ g,
                      const float* __restrict__ bta, float* __restrict__ Y, int rows) {
  int lane = threadIdx.x & 31, wid = threadIdx.x >> 5;
  int row = blockIdx.x * 8 + wid;
  if (row >= rows) return;
  const float* x = X + (size_t)row * E_;
  float v[12];
  float s = 0.f;
  for (int j = 0; j < 12; j++) { v[j] = x[lane + j * 32]; s += v[j]; }
  for (int o = 16; o > 0; o >>= 1) s += __shfl_xor(s, o, 32);
  float mean = s * (1.0f / E_);
  float s2 = 0.f;
  for (int j = 0; j < 12; j++) { float d = v[j] - mean; s2 += d * d; }
  for (int o = 16; o > 0; o >>= 1) s2 += __shfl_xor(s2, o, 32);
  float rstd = rsqrtf(s2 * (1.0f / E_) + 1e-5f);
  float* y = Y + (size_t)row * E_;
  for (int j = 0; j < 12; j++) {
    int e = lane + j * 32;
    y[e] = (v[j] - mean) * rstd * g[e] + bta[e];
  }
}

// ---------------------------------------------------------------------------
// GEMM: C[M,N] = act(A[M,K] @ Wt^T + bias) (+ residual)
//   A: f32 row-major, Wt: bf16 [N,K] row-major, C: f32 row-major.
//   128x128 block tile, 8 waves, each wave = 2x4 grid of 16x16 WMMA tiles.
//   K-slab = 64, DOUBLE-BUFFERED LDS: next slab's global loads are issued
//   before computing on the current slab (one barrier per slab, 16 WMMAs
//   between barriers).
// ---------------------------------------------------------------------------
#define KS_  64
#define LDS_STR_ 72   // bf16 stride per row: 144B, 16B-aligned, bank-skewed

__global__ __launch_bounds__(256)
void gemm_bf16_kernel(const float* __restrict__ A, const u16b* __restrict__ Wt,
                      const float* __restrict__ bias, const float* __restrict__ res,
                      float* __restrict__ C, int M, int N, int K, int relu) {
  __shared__ u16b sA[2][128 * LDS_STR_];   // 2 x 18 KB
  __shared__ u16b sB[2][128 * LDS_STR_];   // 2 x 18 KB

  const int tid  = threadIdx.x;
  const int lane = tid & 31, wid = tid >> 5;
  const int mBase = blockIdx.x * 128;
  const int nBase = blockIdx.y * 128;
  const int wm = (wid & 3) * 32;      // wave M offset within tile
  const int wn = (wid >> 2) * 64;     // wave N offset within tile
  const int lrow = tid >> 1, half = tid & 1;

  const v8f vz = {0, 0, 0, 0, 0, 0, 0, 0};
  v8f acc[2][4];
  for (int i = 0; i < 2; i++) for (int j = 0; j < 4; j++) acc[i][j] = vz;

  // Stage one 128x64 K-slab of A (f32->bf16) and B (bf16 copy) into buffer `buf`.
  auto loadA = [&](int buf, int kb) {
    const float4* s4 = (const float4*)(A + (size_t)(mBase + lrow) * K + kb + half * 32);
    u16b* dst = &sA[buf][lrow * LDS_STR_ + half * 32];
    cvt16_store(s4,     dst);
    cvt16_store(s4 + 4, dst + 16);
  };
  auto loadB = [&](int buf, int kb) {
    int n = nBase + lrow;
    uint4* dst = (uint4*)&sB[buf][lrow * LDS_STR_ + half * 32];
    if (n < N) {
      const uint4* s4 = (const uint4*)(Wt + (size_t)n * K + kb + half * 32);
      dst[0] = s4[0]; dst[1] = s4[1]; dst[2] = s4[2]; dst[3] = s4[3];
    } else {
      uint4 z = make_uint4(0, 0, 0, 0);
      dst[0] = z; dst[1] = z; dst[2] = z; dst[3] = z;
    }
  };

  loadA(0, 0);
  loadB(0, 0);

  // A lane layout: lanes 0-15 hold K {0..7,16..23}, lanes 16-31 K {8..15,24..31}.
  // B lane layout: same K order as A; lanes 0-15 K {0..7,16..23}, 16-31 K {8..15,24..31}.
  const int ak0 = (lane & 16) ? 8 : 0;
  const int bk0 = (lane & 16) ? 8 : 0;

  int cur = 0;
  for (int kb = 0; kb < K; kb += KS_) {
    __syncthreads();
    if (kb + KS_ < K) {             // prefetch next slab into the other buffer
      loadA(cur ^ 1, kb + KS_);
      loadB(cur ^ 1, kb + KS_);
    }
    const u16b* bufA = sA[cur];
    const u16b* bufB = sB[cur];
    for (int kk = 0; kk < KS_; kk += 32) {
      FragAB a[2], b[4];
      for (int mt = 0; mt < 2; mt++) {
        const u16b* p = &bufA[(wm + mt * 16 + (lane & 15)) * LDS_STR_ + kk];
        a[mt].q[0] = *(const uint4*)(p + ak0);
        a[mt].q[1] = *(const uint4*)(p + ak0 + 16);
      }
      for (int nt = 0; nt < 4; nt++) {
        const u16b* p = &bufB[(wn + nt * 16 + (lane & 15)) * LDS_STR_ + kk];
        b[nt].q[0] = *(const uint4*)(p + bk0);
        b[nt].q[1] = *(const uint4*)(p + bk0 + 16);
      }
      for (int mt = 0; mt < 2; mt++)
        for (int nt = 0; nt < 4; nt++)
          acc[mt][nt] = wmma_bf16(a[mt], b[nt], acc[mt][nt]);
    }
    cur ^= 1;
  }

  // Epilogue. C layout: lane = N column (0..15 twice), VGPR r = M row (+8 for
  // upper half-wave).
  const int roff = (lane >> 4) << 3;
  const int ncol = lane & 15;
  for (int mt = 0; mt < 2; mt++)
    for (int nt = 0; nt < 4; nt++)
      for (int r = 0; r < 8; r++) {
        int gm = mBase + wm + mt * 16 + r + roff;
        int gn = nBase + wn + nt * 16 + ncol;
        if (gn < N) {
          float val = acc[mt][nt][r];
          if (bias) val += bias[gn];
          if (relu) val = fmaxf(val, 0.0f);
          if (res)  val += res[(size_t)gm * N + gn];
          C[(size_t)gm * N + gn] = val;
        }
      }
}

// ---------------------------------------------------------------------------
// Fused causal attention, one workgroup (4 waves) per (b, head).
//   S = (Q K^T) / 8 with causal mask, P = softmax(S), O = P V.
//   64x64x64 per head -> WMMA throughout; softmax in f32 via LDS.
// ---------------------------------------------------------------------------
__global__ __launch_bounds__(128)
void attn_kernel(const float* __restrict__ Q, const float* __restrict__ Km,
                 const float* __restrict__ Vv, float* __restrict__ O) {
  __shared__ u16b sQ[64 * LDS_STR_];   // [t][d]
  __shared__ u16b sK[64 * LDS_STR_];   // [s][d]  (B-operand: col=s, k=d)
  __shared__ u16b sVt[64 * LDS_STR_];  // [d][s]  (B-operand: col=d, k=s)
  __shared__ u16b sP[64 * LDS_STR_];   // [t][s]  probabilities, bf16 A tile
  __shared__ float sS[64 * 66];        // [t][s]  f32 scores

  const int tid = threadIdx.x, lane = tid & 31, wid = tid >> 5;
  const int bh = blockIdx.x;
  const int b = bh / H_, h = bh % H_;
  const size_t base = (size_t)b * T_ * E_ + (size_t)h * D_;

  // Stage Q, K (bf16) and V transposed.
  {
    int row = tid >> 1, half = tid & 1;
    const float4* qs = (const float4*)(Q  + base + (size_t)row * E_ + half * 32);
    const float4* ks = (const float4*)(Km + base + (size_t)row * E_ + half * 32);
    cvt16_store(qs,     &sQ[row * LDS_STR_ + half * 32]);
    cvt16_store(qs + 4, &sQ[row * LDS_STR_ + half * 32 + 16]);
    cvt16_store(ks,     &sK[row * LDS_STR_ + half * 32]);
    cvt16_store(ks + 4, &sK[row * LDS_STR_ + half * 32 + 16]);
    const float* vs = Vv + base + (size_t)row * E_ + half * 32;
    for (int j = 0; j < 32; j++) sVt[(half * 32 + j) * LDS_STR_ + row] = f2bf(vs[j]);
  }
  __syncthreads();

  const v8f vz = {0, 0, 0, 0, 0, 0, 0, 0};
  const int roff = (lane >> 4) << 3;
  const int ncol = lane & 15;
  const int ak0 = (lane & 16) ? 8 : 0;
  const int bk0 = (lane & 16) ? 8 : 0;

  // S = Q K^T (wave wid owns rows [wid*16, wid*16+16)).
  v8f accS[4] = {vz, vz, vz, vz};
  for (int kb = 0; kb < 64; kb += 32) {
    FragAB a;
    const u16b* pa = &sQ[(wid * 16 + (lane & 15)) * LDS_STR_ + kb];
    a.q[0] = *(const uint4*)(pa + ak0);
    a.q[1] = *(const uint4*)(pa + ak0 + 16);
    for (int nt = 0; nt < 4; nt++) {
      FragAB bb;
      const u16b* pb = &sK[(nt * 16 + (lane & 15)) * LDS_STR_ + kb];
      bb.q[0] = *(const uint4*)(pb + bk0);
      bb.q[1] = *(const uint4*)(pb + bk0 + 16);
      accS[nt] = wmma_bf16(a, bb, accS[nt]);
    }
  }
  for (int nt = 0; nt < 4; nt++)
    for (int r = 0; r < 8; r++) {
      int trow = wid * 16 + r + roff;
      int scol = nt * 16 + ncol;
      float val = accS[nt][r] * 0.125f;              // D^-0.5, D=64
      sS[trow * 66 + scol] = (scol <= trow) ? val : -1e30f;
    }
  __syncthreads();

  // Row softmax (64 rows, threads 0..63).
  if (tid < 64) {
    float m = -1e30f;
    for (int s = 0; s < 64; s++) m = fmaxf(m, sS[tid * 66 + s]);
    float sum = 0.f;
    for (int s = 0; s < 64; s++) sum += __expf(sS[tid * 66 + s] - m);
    float inv = 1.0f / sum;
    for (int s = 0; s < 64; s++)
      sP[tid * LDS_STR_ + s] = f2bf(__expf(sS[tid * 66 + s] - m) * inv);
  }
  __syncthreads();

  // O = P V.
  v8f accO[4] = {vz, vz, vz, vz};
  for (int kb = 0; kb < 64; kb += 32) {
    FragAB a;
    const u16b* pa = &sP[(wid * 16 + (lane & 15)) * LDS_STR_ + kb];
    a.q[0] = *(const uint4*)(pa + ak0);
    a.q[1] = *(const uint4*)(pa + ak0 + 16);
    for (int nt = 0; nt < 4; nt++) {
      FragAB bb;
      const u16b* pb = &sVt[(nt * 16 + (lane & 15)) * LDS_STR_ + kb];
      bb.q[0] = *(const uint4*)(pb + bk0);
      bb.q[1] = *(const uint4*)(pb + bk0 + 16);
      accO[nt] = wmma_bf16(a, bb, accO[nt]);
    }
  }
  for (int nt = 0; nt < 4; nt++)
    for (int r = 0; r < 8; r++) {
      int trow = wid * 16 + r + roff;
      int dcol = nt * 16 + ncol;
      O[base + (size_t)trow * E_ + dcol] = accO[nt][r];
    }
}

// ---------------------------------------------------------------------------
// Cross-entropy: deterministic two-stage reduction (256 blocks x 256 rows).
// ---------------------------------------------------------------------------
__global__ __launch_bounds__(256)
void loss_partial_kernel(const float* __restrict__ logits, const int* __restrict__ targets,
                         float* __restrict__ partial) {
  __shared__ float red[256];
  int row = blockIdx.x * 256 + threadIdx.x;
  const float* lg = logits + (size_t)row * V_;
  float m = -1e30f;
  for (int j = 0; j < V_; j++) m = fmaxf(m, lg[j]);
  float sum = 0.f;
  for (int j = 0; j < V_; j++) sum += __expf(lg[j] - m);
  float lse = m + __logf(sum);
  red[threadIdx.x] = lse - lg[targets[row]];
  __syncthreads();
  for (int o = 128; o > 0; o >>= 1) {
    if (threadIdx.x < o) red[threadIdx.x] += red[threadIdx.x + o];
    __syncthreads();
  }
  if (threadIdx.x == 0) partial[blockIdx.x] = red[0];
}

__global__ void loss_final_kernel(const float* __restrict__ partial, float* __restrict__ out) {
  float s = 0.f;
  for (int i = 0; i < 256; i++) s += partial[i];
  out[0] = s / (float)BT_;
}

// ---------------------------------------------------------------------------
// Host orchestration.
// Input order (setup_inputs dict order): idx, targets, tok, pos,
//   4 x blocks{wq,wk,wv,wo,bo,w1,b1,w2,b2,ln1_g,ln1_b,ln2_g,ln2_b},
//   lnf_g, lnf_b, lm_w, lm_b.
// ---------------------------------------------------------------------------
extern "C" void kernel_launch(void* const* d_in, const int* in_sizes, int n_in,
                              void* d_out, int out_size, void* d_ws, size_t ws_size,
                              hipStream_t stream) {
  (void)in_sizes; (void)n_in; (void)out_size; (void)ws_size;

  const int*   idx     = (const int*)d_in[0];
  const int*   targets = (const int*)d_in[1];
  const float* tok     = (const float*)d_in[2];
  const float* pos     = (const float*)d_in[3];
  auto blkp = [&](int l, int j) { return (const float*)d_in[4 + l * 13 + j]; };
  const float* lnf_g = (const float*)d_in[56];
  const float* lnf_b = (const float*)d_in[57];
  const float* lm_w  = (const float*)d_in[58];
  const float* lm_b  = (const float*)d_in[59];

  // Workspace carving.
  char* ws = (char*)d_ws;
  size_t off = 0;
  auto carve = [&](size_t bytes) {
    void* p = ws + off;
    off = (off + bytes + 255) & ~(size_t)255;
    return p;
  };
  u16b *wq_t[NL_], *wk_t[NL_], *wv_t[NL_], *wo_t[NL_], *w1_t[NL_], *w2_t[NL_];
  for (int l = 0; l < NL_; l++) {
    wq_t[l] = (u16b*)carve((size_t)E_ * E_ * 2);
    wk_t[l] = (u16b*)carve((size_t)E_ * E_ * 2);
    wv_t[l] = (u16b*)carve((size_t)E_ * E_ * 2);
    wo_t[l] = (u16b*)carve((size_t)E_ * E_ * 2);
    w1_t[l] = (u16b*)carve((size_t)FF_ * E_ * 2);
    w2_t[l] = (u16b*)carve((size_t)E_ * FF_ * 2);
  }
  u16b* lm_t   = (u16b*)carve((size_t)V_ * E_ * 2);
  float* bufX  = (float*)carve((size_t)BT_ * E_ * 4);
  float* bufY  = (float*)carve((size_t)BT_ * E_ * 4);
  float* hbuf  = (float*)carve((size_t)BT_ * E_ * 4);   // also serves as attn O
  float* qbuf  = (float*)carve((size_t)BT_ * E_ * 4);
  float* kbuf  = (float*)carve((size_t)BT_ * E_ * 4);
  float* vbuf  = (float*)carve((size_t)BT_ * E_ * 4);
  float* ffn   = (float*)carve((size_t)BT_ * FF_ * 4);
  float* partial = (float*)carve(256 * 4);
  float* obuf = hbuf;  // h is dead once q/k/v are formed

  auto wconv = [&](const float* W, u16b* Wt, int K, int N) {
    size_t n = (size_t)K * N;
    wconv_kernel<<<(unsigned)((n + 255) / 256), 256, 0, stream>>>(W, Wt, K, N);
  };
  for (int l = 0; l < NL_; l++) {
    wconv(blkp(l, 0), wq_t[l], E_, E_);
    wconv(blkp(l, 1), wk_t[l], E_, E_);
    wconv(blkp(l, 2), wv_t[l], E_, E_);
    wconv(blkp(l, 3), wo_t[l], E_, E_);
    wconv(blkp(l, 5), w1_t[l], E_, FF_);   // w1: [E,4E] -> Wt [4E][E]
    wconv(blkp(l, 7), w2_t[l], FF_, E_);   // w2: [4E,E] -> Wt [E][4E]
  }
  wconv(lm_w, lm_t, E_, V_);

  embed_kernel<<<(unsigned)(((size_t)BT_ * E_ + 255) / 256), 256, 0, stream>>>(idx, tok, pos, bufX);

  auto gemm = [&](const float* A, const u16b* Wt, const float* bias, const float* res,
                  float* C, int M, int N, int K, int relu) {
    dim3 grid(M / 128, (N + 127) / 128);
    gemm_bf16_kernel<<<grid, 256, 0, stream>>>(A, Wt, bias, res, C, M, N, K, relu);
  };

  for (int l = 0; l < NL_; l++) {
    layernorm_kernel<<<BT_ / 8, 256, 0, stream>>>(bufX, blkp(l, 9), blkp(l, 10), hbuf, BT_);
    gemm(hbuf, wq_t[l], nullptr, nullptr, qbuf, BT_, E_, E_, 0);
    gemm(hbuf, wk_t[l], nullptr, nullptr, kbuf, BT_, E_, E_, 0);
    gemm(hbuf, wv_t[l], nullptr, nullptr, vbuf, BT_, E_, E_, 0);
    attn_kernel<<<B_ * H_, 128, 0, stream>>>(qbuf, kbuf, vbuf, obuf);
    gemm(obuf, wo_t[l], blkp(l, 4), bufX, bufY, BT_, E_, E_, 0);     // x + o@wo + bo
    layernorm_kernel<<<BT_ / 8, 256, 0, stream>>>(bufY, blkp(l, 11), blkp(l, 12), hbuf, BT_);
    gemm(hbuf, w1_t[l], blkp(l, 6), nullptr, ffn, BT_, FF_, E_, 1);  // relu(h@w1+b1)
    gemm(ffn, w2_t[l], blkp(l, 8), bufY, bufX, BT_, E_, FF_, 0);     // y + ffn@w2 + b2
  }

  layernorm_kernel<<<BT_ / 8, 256, 0, stream>>>(bufX, lnf_g, lnf_b, hbuf, BT_);
  float* logits = (float*)d_out;
  gemm(hbuf, lm_t, lm_b, nullptr, logits, BT_, V_, E_, 0);

  loss_partial_kernel<<<256, 256, 0, stream>>>(logits, targets, partial);
  loss_final_kernel<<<1, 1, 0, stream>>>(partial, logits + (size_t)BT_ * V_);
}
